// VQVAE2Test_16080357556667
// MI455X (gfx1250) — matrix operand order repaired and checked by reading, not compiled, not run on hardware
//
#include <hip/hip_runtime.h>
#include <hip/hip_bf16.h>

// ---------------------------------------------------------------------------
// VQ-VAE-2 forward for MI455X (gfx1250, wave32).
//  - convs/tconvs: implicit GEMM, v_wmma_f32_16x16x32_f16, T co-tiles per wave
//  - BN(training)+LeakyReLU folded into consumers as per-channel (a,b,slope)
//  - weight tiles staged into LDS via Tensor Data Mover (tensor_load_to_lds)
//  - BN stats: wave butterfly-reduce -> LDS atomics -> 1 global atomic/chan/blk
//  - VQ nearest-code: WMMA GEMM + cross-lane min reduce
// ---------------------------------------------------------------------------

typedef _Float16 half_t;
typedef __attribute__((ext_vector_type(16))) _Float16     v16h;
typedef __attribute__((ext_vector_type(8)))  float        v8f;
typedef __attribute__((ext_vector_type(4)))  unsigned int u32x4;
typedef __attribute__((ext_vector_type(8)))  int          i32x8;
typedef __attribute__((ext_vector_type(4)))  int          i32x4;

#define WAVE 32
#define WPB  8
#define BLOCK (WAVE * WPB)

#define HAS_TDM (__has_builtin(__builtin_amdgcn_tensor_load_to_lds) && \
                 __has_builtin(__builtin_amdgcn_s_wait_tensorcnt))

// ---------------------------------------------------------------------------
// Implicit-GEMM conv / tconv. T = number of 16-row co tiles (CoPad = 16*T).
// ---------------------------------------------------------------------------
template <int T>
__global__ __launch_bounds__(BLOCK)
void conv_wmma_kernel(const half_t* __restrict__ act,
                      const half_t* __restrict__ wpk,
                      const float*  __restrict__ bias,
                      const float*  __restrict__ ta,   // input BN scale  (a)
                      const float*  __restrict__ tb,   // input BN shift  (b)
                      const float*  __restrict__ ts,   // input lrelu slope
                      float*        __restrict__ stats,
                      half_t*       __restrict__ outh,
                      float*        __restrict__ outf,
                      int N, int IH, int IW, int Ci,
                      int OH, int OW, int Co, int Kp,
                      int stride, int tmode,
                      int outCstride, int outCoff, int mode)
{
    extern __shared__ half_t sA[];                 // T*16 rows x Kp halves
    __shared__ float sSum[64], sSq[64];
    __shared__ float sTa[128], sTb[128], sTs[128];

    const int lane = threadIdx.x & 31;
    const int wave = threadIdx.x >> 5;

    // stage input transform + zero block stats
    for (int c = threadIdx.x; c < Ci; c += BLOCK) {
        sTa[c] = ta[c]; sTb[c] = tb[c]; sTs[c] = ts[c];
    }
    if (threadIdx.x < 64) { sSum[threadIdx.x] = 0.f; sSq[threadIdx.x] = 0.f; }

    // ---- stage packed weight tile [T*16 x Kp] into LDS ----
    const int nW = T * 16 * Kp;
#if HAS_TDM
    if (wave == 0) {
        const unsigned long long ga = (unsigned long long)(uintptr_t)wpk;
        const unsigned ldsa = (unsigned)(uintptr_t)sA;    // low 32 = LDS offset
        u32x4 g0;
        g0[0] = 1u;                                       // count=1, user mode
        g0[1] = ldsa;                                     // lds_addr
        g0[2] = (unsigned)(ga & 0xffffffffull);           // global_addr[31:0]
        g0[3] = (unsigned)((ga >> 32) & 0x1ffffffull) | (2u << 30); // type=2
        const unsigned td0 = (unsigned)Kp;                // tensor_dim0
        const unsigned td1 = (unsigned)(T * 16);          // tensor_dim1
        i32x8 g1;
        g1[0] = (int)(1u << 16);                          // data_size=1 (2B)
        g1[1] = (int)((td0 & 0xffffu) << 16);             // tensor_dim0 lo
        g1[2] = (int)((td0 >> 16) | ((td1 & 0xffffu) << 16));
        g1[3] = (int)((td1 >> 16) | ((td0 & 0xffffu) << 16)); // tile_dim0=Kp
        g1[4] = (int)(td1 & 0xffffu);                     // tile_dim1, tile_dim2=0
        g1[5] = (int)td0;                                 // dim0_stride lo (=Kp)
        g1[6] = 0;                                        // dim0_stride hi
        g1[7] = 0;
        i32x4 gz4 = {0, 0, 0, 0};
        i32x8 gz8 = {0, 0, 0, 0, 0, 0, 0, 0};
        __builtin_amdgcn_tensor_load_to_lds(g0, g1, gz4, gz4, gz8, 0);
        __builtin_amdgcn_s_wait_tensorcnt(0);
    }
#else
    for (int i = threadIdx.x; i < nW; i += BLOCK) sA[i] = wpk[i];
#endif
    __syncthreads();

    const int npix = N * OH * OW;
    const int tile = blockIdx.x * WPB + wave;       // wave-uniform pixel tile
    const bool wactive = (tile * 16 < npix);

    if (wactive) {
        const int col = lane & 15;
        const int hi  = lane >> 4;
        const int pix = tile * 16 + col;            // B column / output pixel
        const int ow  = pix % OW;
        const int t0  = pix / OW;
        const int oh  = t0 % OH;
        const int n   = t0 / OH;

        auto cellSetup = [&](int cell, size_t& base) -> bool {
            if (cell >= 9) return false;
            const int kh = cell / 3, kw = cell - kh * 3;
            int iy, ix; bool ok;
            if (!tmode) {
                iy = oh * stride - 1 + kh;
                ix = ow * stride - 1 + kw;
                ok = (iy >= 0) & (iy < IH) & (ix >= 0) & (ix < IW);
            } else {
                const int ty = oh + 1 - kh, tx = ow + 1 - kw;
                iy = ty >> 1; ix = tx >> 1;
                ok = (ty >= 0) & ((ty & 1) == 0) & (iy < IH) &
                     (tx >= 0) & ((tx & 1) == 0) & (ix < IW);
            }
            if (ok) base = (((size_t)n * IH + iy) * IW + ix) * Ci;
            return ok;
        };

        v8f acc[T];
#pragma unroll
        for (int t = 0; t < T; ++t) acc[t] = (v8f){};

        for (int k0 = 0; k0 < Kp; k0 += 32) {
            // ---- B fragment: 16 consecutive K per lane, incremental walk ----
            const int kk = k0 + hi * 16;
            int cell = kk / Ci;                     // one divide per chunk
            int c    = kk - cell * Ci;
            size_t base = 0;
            bool okc = cellSetup(cell, base);
            v16h b;
#pragma unroll
            for (int u = 0; u < 16; ++u) {
                half_t val = (half_t)0.f;
                if (okc) {
                    float x = (float)act[base + c];
                    x = sTa[c] * x + sTb[c];        // BN affine
                    val = (half_t)(x > 0.f ? x : sTs[c] * x);  // leaky relu
                }
                b[u] = val;
                if (++c == Ci) { c = 0; ++cell; okc = cellSetup(cell, base); }
            }
            // ---- A fragments from LDS (b32 pairs), one wmma per co tile ----
#pragma unroll
            for (int t = 0; t < T; ++t) {
                union { v16h h; unsigned int u[8]; } a;
#pragma unroll
                for (int v = 0; v < 8; ++v) {
                    const int kb = hi * 16 + 2 * v;
                    a.u[v] = *(const unsigned int*)&sA[(size_t)(t * 16 + col) * Kp + k0 + kb];
                }
                acc[t] = __builtin_amdgcn_wmma_f32_16x16x32_f16(
                    false, a.h, false, b, (short)0, acc[t], false, false);
            }
        }

        // ---- epilogue ----
#pragma unroll
        for (int t = 0; t < T; ++t) {
#pragma unroll
            for (int j = 0; j < 8; ++j) {
                const int co = t * 16 + j + hi * 8;
                const bool live = (co < Co);
                const float v = acc[t][j] + (live ? bias[co] : 0.f);
                if (live) {
                    if (mode == 0) {
                        outh[(size_t)pix * outCstride + outCoff + co] = (half_t)v;
                    } else {
                        const float s = 1.f / (1.f + __expf(-v));
                        outf[(((size_t)n * Co + co) * OH + oh) * OW + ow] = s;
                    }
                }
                if (mode == 0) {
                    float s1 = live ? v : 0.f;
                    float s2 = s1 * s1;
#pragma unroll
                    for (int off = 1; off < 16; off <<= 1) {
                        s1 += __shfl_xor(s1, off, 32);
                        s2 += __shfl_xor(s2, off, 32);
                    }
                    if (col == 0 && live) {         // lanes 0 and 16
                        atomicAdd(&sSum[co], s1);   // ds_add_f32
                        atomicAdd(&sSq[co],  s2);
                    }
                }
            }
        }
    }

    __syncthreads();
    if (mode == 0) {
        for (int c = threadIdx.x; c < Co; c += BLOCK) {
            atomicAdd(&stats[c],       sSum[c]);
            atomicAdd(&stats[256 + c], sSq[c]);
        }
    }
}

// ---------------------------------------------------------------------------
// VQ nearest-code. Input z is pre-BN; (ta,tb) + slope 0.1 applied inline.
// ---------------------------------------------------------------------------
__global__ __launch_bounds__(BLOCK)
void vq_wmma_kernel(const half_t* __restrict__ z,
                    const float*  __restrict__ emb,
                    const float*  __restrict__ ee,
                    const float*  __restrict__ ta,
                    const float*  __restrict__ tb,
                    int Kcodes, int C, int npix,
                    half_t* __restrict__ q, int qCstride, int qCoff)
{
    const int lane = threadIdx.x & 31;
    const int wave = threadIdx.x >> 5;
    const int tile = blockIdx.x * WPB + wave;
    if (tile * 16 >= npix) return;
    const int col = lane & 15;
    const int hi  = lane >> 4;
    const int apx = tile * 16 + col;

    auto ldz = [&](int c) -> half_t {
        float x = (float)z[(size_t)apx * C + c];
        x = ta[c] * x + tb[c];
        return (half_t)(x > 0.f ? x : 0.1f * x);
    };

    v16h a0, a1;
#pragma unroll
    for (int v = 0; v < 8; ++v) {
        const int kb = hi * 16 + 2 * v;
        a0[2 * v]     = ldz(kb);       a0[2 * v + 1] = ldz(kb + 1);
        a1[2 * v]     = ldz(32 + kb);  a1[2 * v + 1] = ldz(32 + kb + 1);
    }

    float best[8]; int bidx[8];
#pragma unroll
    for (int j = 0; j < 8; ++j) { best[j] = 3.4e38f; bidx[j] = 0; }

    for (int t = 0; t < Kcodes / 16; ++t) {
        const int code = t * 16 + col;
        v16h b0, b1;
#pragma unroll
        for (int v = 0; v < 8; ++v) {
            const int kr = hi * 16 + 2 * v;
            b0[2 * v]     = (half_t)emb[(size_t)code * C + kr];
            b0[2 * v + 1] = (half_t)emb[(size_t)code * C + kr + 1];
            b1[2 * v]     = (half_t)emb[(size_t)code * C + 32 + kr];
            b1[2 * v + 1] = (half_t)emb[(size_t)code * C + 32 + kr + 1];
        }
        v8f s = {};
        s = __builtin_amdgcn_wmma_f32_16x16x32_f16(false, a0, false, b0,
                                                   (short)0, s, false, false);
        s = __builtin_amdgcn_wmma_f32_16x16x32_f16(false, a1, false, b1,
                                                   (short)0, s, false, false);
        const float ec = ee[code];
#pragma unroll
        for (int j = 0; j < 8; ++j) {
            const float d = ec - 2.f * s[j];
            if (d < best[j]) { best[j] = d; bidx[j] = code; }
        }
    }

#pragma unroll
    for (int off = 1; off < 16; off <<= 1) {
#pragma unroll
        for (int j = 0; j < 8; ++j) {
            const float od = __shfl_xor(best[j], off, 32);
            const int   oi = __shfl_xor(bidx[j], off, 32);
            if (od < best[j] || (od == best[j] && oi < bidx[j])) {
                best[j] = od; bidx[j] = oi;
            }
        }
    }

#pragma unroll
    for (int j = 0; j < 8; ++j) {
        const int p   = tile * 16 + hi * 8 + j;
        const int idx = bidx[j];
        for (int c = col; c < C; c += 16)
            q[(size_t)p * qCstride + qCoff + c] = (half_t)emb[(size_t)idx * C + c];
    }
}

// ---------------------------------------------------------------------------
// Utility kernels
// ---------------------------------------------------------------------------
__global__ void pack_w_kernel(const float* __restrict__ w, half_t* __restrict__ wpk,
                              int Ci, int Co, int CoPad, int Kp, int tmode)
{
    const int i = blockIdx.x * 256 + threadIdx.x;
    if (i >= CoPad * Kp) return;
    const int co = i / Kp, K = i - co * Kp;
    float v = 0.f;
    if (co < Co && K < 9 * Ci) {
        const int cell = K / Ci, c = K - cell * Ci;
        const int kh = cell / 3, kw = cell - kh * 3;
        v = tmode ? w[(((size_t)c * Co + co) * 3 + kh) * 3 + kw]   // (Ci,Co,3,3)
                  : w[(((size_t)co * Ci + c) * 3 + kh) * 3 + kw];  // (Co,Ci,3,3)
    }
    wpk[i] = (half_t)v;
}

// stats slot: [0]sum [256]sumsq [512]a [768]b [1024]slope  (1280 floats)
__global__ void finalize_stats_kernel(float* st, const float* __restrict__ g,
                                      const float* __restrict__ bt,
                                      int Co, float inv_m)
{
    const int c = threadIdx.x;
    if (c >= Co) return;
    const float mean = st[c] * inv_m;
    const float var  = st[256 + c] * inv_m - mean * mean;
    const float rstd = rsqrtf(var + 1e-5f);
    const float a    = g[c] * rstd;
    st[512 + c]  = a;
    st[768 + c]  = bt[c] - a * mean;
    st[1024 + c] = 0.1f;
}

__global__ void fill_ident_kernel(float* t)     // a[256]=1, b[256]=0, s[256]=1
{
    const int i = blockIdx.x * 256 + threadIdx.x;
    if (i < 256)      t[i] = 1.f;
    else if (i < 512) t[i] = 0.f;
    else if (i < 768) t[i] = 1.f;
}

// d1 input transform: c<64 -> 'up' BN+lrelu, c>=64 -> identity (qb)
__global__ void fill_d1t_kernel(const float* __restrict__ upst, float* t)
{
    const int c = threadIdx.x;
    if (c >= 128) return;
    const bool u = c < 64;
    t[c]       = u ? upst[512 + c] : 1.f;   // a
    t[128 + c] = u ? upst[768 + c] : 0.f;   // b
    t[256 + c] = u ? 0.1f : 1.f;            // slope
}

__global__ void x_to_nhwc_kernel(const float* __restrict__ x, half_t* __restrict__ out,
                                 int N, int C, int H, int W)
{
    const int i = blockIdx.x * 256 + threadIdx.x;
    if (i >= N * C * H * W) return;
    const int c = i % C;
    const int w = (i / C) % W;
    const int h = (i / (C * W)) % H;
    const int n = i / (C * W * H);
    out[i] = (half_t)x[(((size_t)n * C + c) * H + h) * W + w];
}

__global__ void ee_kernel(const float* __restrict__ emb, float* __restrict__ ee,
                          int K, int C)
{
    const int k = blockIdx.x * 256 + threadIdx.x;
    if (k >= K) return;
    float s = 0.f;
    for (int c = 0; c < C; ++c) { const float v = emb[(size_t)k * C + c]; s += v * v; }
    ee[k] = s;
}

// ---------------------------------------------------------------------------
// Host-side orchestration
// ---------------------------------------------------------------------------
struct LayerDesc {
    int wi;                        // d_in index of weight (b=+1, g=+2, bt=+3)
    int Ci, Co, Kp;
    int IH, IW, OH, OW, stride, tmode;
    const half_t* in; half_t* outh; float* outf;
    int outCstride, outCoff, mode;
    int slot;                      // this layer's stats slot
    const float* ta; const float* tb; const float* ts;  // input transform
};

extern "C" void kernel_launch(void* const* d_in, const int* in_sizes, int n_in,
                              void* d_out, int out_size, void* d_ws, size_t ws_size,
                              hipStream_t stream)
{
    (void)in_sizes; (void)n_in; (void)out_size; (void)ws_size;
    const int N = 32;

    // ---- workspace arena ----
    char* p = (char*)d_ws;
    half_t* A0  = (half_t*)p; p += 67108864;     // ping (max 32x256x256x16 f16)
    half_t* A1  = (half_t*)p; p += 67108864;     // pong
    half_t* CC  = (half_t*)p; p += 8388608;      // concat [32,32,32,128] f16
    half_t* ZT  = (half_t*)p; p += 1048576;      // zt [32,16,16,64] f16
    half_t* QT  = (half_t*)p; p += 1048576;      // quantized top
    half_t* WPK = (half_t*)p; p += 1048576;      // packed f16 weights
    float*  STATS = (float*)p; p += 12 * 1280 * 4;  // 12 slots
    float*  EE    = (float*)p; p += 4096;        // ||emb||^2 top/bottom
    float*  IDT   = (float*)p; p += 768 * 4;     // identity transform
    float*  D1T   = (float*)p; p += 384 * 4;     // d1 mixed transform

    const float* xin   = (const float*)d_in[0];
    const float* emb_t = (const float*)d_in[47];
    const float* emb_b = (const float*)d_in[48];

    (void)hipMemsetAsync(STATS, 0, 12 * 1280 * 4, stream);
    fill_ident_kernel<<<3, 256, 0, stream>>>(IDT);
    {
        const int tot = N * 3 * 256 * 256;
        x_to_nhwc_kernel<<<(tot + 255) / 256, 256, 0, stream>>>(xin, A0, N, 3, 256, 256);
    }
    ee_kernel<<<2, 256, 0, stream>>>(emb_t, EE,       512, 64);
    ee_kernel<<<2, 256, 0, stream>>>(emb_b, EE + 512, 512, 64);

    auto slotA = [&](int s) { return STATS + (size_t)s * 1280 + 512; };
    auto slotB = [&](int s) { return STATS + (size_t)s * 1280 + 768; };
    auto slotS = [&](int s) { return STATS + (size_t)s * 1280 + 1024; };

    LayerDesc L[12] = {
      //  wi  Ci  Co   Kp   IH  IW  OH  OW  s  t  in  outh  outf str off md slot  ta tb ts
      {  1,   3,   8,  32, 256,256,256,256, 1, 0, A0, A1, nullptr,   8, 0, 0, 0, IDT, IDT+256, IDT+512 },
      {  5,   8,  16,  96, 256,256,256,256, 1, 0, A1, A0, nullptr,  16, 0, 0, 1, slotA(0), slotB(0), slotS(0) },
      {  9,  16,  24, 160, 256,256,128,128, 2, 0, A0, A1, nullptr,  24, 0, 0, 2, slotA(1), slotB(1), slotS(1) },
      { 13,  24,  32, 224, 128,128, 64, 64, 2, 0, A1, A0, nullptr,  32, 0, 0, 3, slotA(2), slotB(2), slotS(2) },
      { 17,  32,  64, 288,  64, 64, 32, 32, 2, 0, A0, A1, nullptr,  64, 0, 0, 4, slotA(3), slotB(3), slotS(3) },
      { 21,  64,  56, 576,  32, 32, 16, 16, 2, 0, A1, A0, nullptr,  56, 0, 0, 5, slotA(4), slotB(4), slotS(4) },
      { 25,  56,  64, 512,  16, 16, 16, 16, 1, 0, A0, ZT, nullptr,  64, 0, 0, 6, slotA(5), slotB(5), slotS(5) },
      { 29,  64,  64, 576,  16, 16, 32, 32, 2, 1, QT, CC, nullptr, 128, 0, 0, 7, IDT, IDT+256, IDT+512 },
      { 33, 128,  32,1152,  32, 32, 64, 64, 2, 1, CC, A0, nullptr,  32, 0, 0, 8, D1T, D1T+128, D1T+256 },
      { 37,  32,  24, 288,  64, 64,128,128, 2, 1, A0, A1, nullptr,  24, 0, 0, 9, slotA(8), slotB(8), slotS(8) },
      { 41,  24,  16, 224, 128,128,256,256, 2, 1, A1, A0, nullptr,  16, 0, 0,10, slotA(9), slotB(9), slotS(9) },
      { 45,  16,   3, 160, 256,256,256,256, 1, 0, A0, nullptr, (float*)d_out,
                                                               0, 0, 1,11, slotA(10), slotB(10), slotS(10) },
    };

    size_t woff[12];
    {
        size_t acc = 0;
        for (int i = 0; i < 12; ++i) {
            const int CoPad = ((L[i].Co + 15) / 16) * 16;
            woff[i] = acc;
            acc += (size_t)CoPad * L[i].Kp;
        }
    }

    auto run_layer = [&](int i) {
        const LayerDesc& l = L[i];
        const int T = (l.Co + 15) / 16;
        const int CoPad = T * 16;
        half_t* wpk = WPK + woff[i];
        const float* w    = (const float*)d_in[l.wi];
        const float* bias = (const float*)d_in[l.wi + 1];
        float* st = STATS + (size_t)l.slot * 1280;

        {
            const int tot = CoPad * l.Kp;
            pack_w_kernel<<<(tot + 255) / 256, 256, 0, stream>>>(
                w, wpk, l.Ci, l.Co, CoPad, l.Kp, l.tmode);
        }
        const int npix  = N * l.OH * l.OW;
        const int tiles = npix / 16;
        const dim3 grid((tiles + WPB - 1) / WPB);
        const size_t smem = (size_t)CoPad * l.Kp * sizeof(half_t);
        switch (T) {
        case 1:
            conv_wmma_kernel<1><<<grid, BLOCK, smem, stream>>>(
                l.in, wpk, bias, l.ta, l.tb, l.ts, st, l.outh, l.outf,
                N, l.IH, l.IW, l.Ci, l.OH, l.OW, l.Co, l.Kp,
                l.stride, l.tmode, l.outCstride, l.outCoff, l.mode);
            break;
        case 2:
            conv_wmma_kernel<2><<<grid, BLOCK, smem, stream>>>(
                l.in, wpk, bias, l.ta, l.tb, l.ts, st, l.outh, l.outf,
                N, l.IH, l.IW, l.Ci, l.OH, l.OW, l.Co, l.Kp,
                l.stride, l.tmode, l.outCstride, l.outCoff, l.mode);
            break;
        default:
            conv_wmma_kernel<4><<<grid, BLOCK, smem, stream>>>(
                l.in, wpk, bias, l.ta, l.tb, l.ts, st, l.outh, l.outf,
                N, l.IH, l.IW, l.Ci, l.OH, l.OW, l.Co, l.Kp,
                l.stride, l.tmode, l.outCstride, l.outCoff, l.mode);
            break;
        }
        if (l.mode == 0) {
            const float* g  = (const float*)d_in[l.wi + 2];
            const float* bt = (const float*)d_in[l.wi + 3];
            finalize_stats_kernel<<<1, 256, 0, stream>>>(st, g, bt, l.Co, 1.f / (float)npix);
        }
    };

    // encoder
    for (int i = 0; i <= 6; ++i) run_layer(i);

    // VQ top: ZT (pre-BN, et2 transform) -> QT
    {
        const int npix = N * 16 * 16;
        vq_wmma_kernel<<<npix / 16 / WPB, BLOCK, 0, stream>>>(
            ZT, emb_t, EE, slotA(6), slotB(6), 512, 64, npix, QT, 64, 0);
    }
    // VQ bottom: A1 (zb pre-BN, eb3 transform) -> CC channels 64..127
    {
        const int npix = N * 32 * 32;
        vq_wmma_kernel<<<npix / 16 / WPB, BLOCK, 0, stream>>>(
            A1, emb_b, EE + 512, slotA(4), slotB(4), 512, 64, npix, CC, 128, 64);
    }

    // decoder: up (pre-BN into CC c0..63), its stats, then d1 transform fill
    run_layer(7);
    fill_d1t_kernel<<<1, 128, 0, stream>>>(STATS + 7 * 1280, D1T);
    for (int i = 8; i <= 11; ++i) run_layer(i);
}
